// DynamicConvolution_7387343749866
// MI455X (gfx1250) — matrix operand order repaired and checked by reading, not compiled, not run on hardware
//
#include <hip/hip_runtime.h>
#include <hip/hip_bf16.h>

typedef __attribute__((ext_vector_type(16))) __bf16 v16bf;
typedef __attribute__((ext_vector_type(8)))  __bf16 v8bf;
typedef __attribute__((ext_vector_type(8)))  float  v8f;

// f32 -> bf16 round-to-nearest-even (bit trick)
static __device__ __forceinline__ __bf16 f2bf(float f) {
    unsigned u = __builtin_bit_cast(unsigned, f);
    unsigned r = u + 0x7FFFu + ((u >> 16) & 1u);
    unsigned short h = (unsigned short)(r >> 16);
    return __builtin_bit_cast(__bf16, h);
}

#define CIN    128
#define COUT   128
#define HH     64
#define WW     64
#define KK     3
#define CICHUNK 32
#define NCHUNK (CIN / CICHUNK)   // 4
#define PADW   (WW + 2)          // 66
#define BS_ELEMS (KK * PADW * CICHUNK)          // 3*66*32 = 6336
#define AS_ELEMS (KK * KK * COUT * CICHUNK)     // 9*128*32 = 36864

__global__ __launch_bounds__(256)
void dynconv_wmma_bf16_kernel(const float* __restrict__ feat,
                              const float* __restrict__ dynk,
                              float* __restrict__ out) {
    __shared__ __bf16 As[AS_ELEMS];  // [khkw][co][ci]   72 KB
    __shared__ __bf16 Bs[BS_ELEMS];  // [row][padcol][ci] 12.7 KB

    const int b   = blockIdx.y;
    const int h   = blockIdx.x;
    const int tid = threadIdx.x;

    const float* X  = feat + b * (CIN * HH * WW);        // [ci][y][x]
    const float* Wt = dynk + b * (COUT * CIN * KK * KK); // [co][ci][kh][kw]
    float*       O  = out  + b * (COUT * HH * WW) + h * WW;

    const int wave    = tid >> 5;
    const int lane    = tid & 31;
    const int co_base = wave * 16;
    const int halfsel = lane >> 4;        // 0 for lanes 0-15, 1 for 16-31
    const int arow    = lane & 15;        // A fragment M row within tile
    const int ahalf   = halfsel * 8;      // A: K base 0 or 8
    const int bhalf   = halfsel * 8;      // B: K base 0 or 8
    const int ncol    = lane & 15;        // B/C column within tile

    v8f acc[4] = {};                      // 4 N-subtiles of 16 cols

    for (int cc = 0; cc < NCHUNK; ++cc) {
        const int ci0 = cc * CICHUNK;
        __syncthreads();   // previous compute done before LDS overwrite

        // ---- stage shifted input rows (f32 -> bf16, zero padded) ----
        for (int idx = tid; idx < BS_ELEMS; idx += 256) {
            int c    = idx % PADW;        // padded col 0..65 -> image col c-1
            int pair = idx / PADW;
            int r    = pair % KK;         // image row h + r - 1
            int ci   = pair / KK;         // 0..31
            int gy   = h + r - 1;
            int gx   = c - 1;
            float v = 0.0f;
            if ((unsigned)gy < (unsigned)HH && (unsigned)gx < (unsigned)WW)
                v = X[(ci0 + ci) * (HH * WW) + gy * WW + gx];
            Bs[(r * PADW + c) * CICHUNK + ci] = f2bf(v);
        }

        // ---- stage weights for all 9 taps of this ci chunk ----
        for (int idx = tid; idx < AS_ELEMS; idx += 256) {
            int ci   = idx & (CICHUNK - 1);
            int t    = idx >> 5;
            int khkw = t % (KK * KK);
            int co   = t / (KK * KK);
            float v = Wt[co * (CIN * KK * KK) + (ci0 + ci) * (KK * KK) + khkw];
            As[(khkw * COUT + co) * CICHUNK + ci] = f2bf(v);
        }
        __syncthreads();

        // ---- compute: 9 taps x 4 N-subtiles of WMMA ----
        for (int khkw = 0; khkw < KK * KK; ++khkw) {
            const int kh = khkw / KK;
            const int kw = khkw % KK;

            // A fragment: 16x32 bf16, documented 16-bit A layout
            const __bf16* ab = &As[((khkw * COUT) + co_base + arow) * CICHUNK + ahalf];
            v8bf alo = *(const v8bf*)(ab);        // K ahalf..ahalf+7
            v8bf ahi = *(const v8bf*)(ab + 16);   // K ahalf+16..ahalf+23
            v16bf afrag = __builtin_shufflevector(alo, ahi,
                0,1,2,3,4,5,6,7,8,9,10,11,12,13,14,15);

            #pragma unroll
            for (int nt = 0; nt < 4; ++nt) {
                const int w = nt * 16 + ncol;
                const __bf16* bb = &Bs[(kh * PADW + w + kw) * CICHUNK + bhalf];
                v8bf blo = *(const v8bf*)(bb);      // K bhalf..bhalf+7
                v8bf bhi = *(const v8bf*)(bb + 16); // K bhalf+16..bhalf+23
                v16bf bfrag = __builtin_shufflevector(blo, bhi,
                    0,1,2,3,4,5,6,7,8,9,10,11,12,13,14,15);

                acc[nt] = __builtin_amdgcn_wmma_f32_16x16x32_bf16(
                    /*neg_a=*/false, afrag, /*neg_b=*/false, bfrag,
                    /*c_mod=*/(short)0, acc[nt],
                    /*reuse_a=*/false, /*reuse_b=*/false);
            }
        }
    }

    // ---- epilogue: C/D layout -> global ----
    const int rowoff = halfsel * 8;
    #pragma unroll
    for (int nt = 0; nt < 4; ++nt) {
        const int w = nt * 16 + ncol;
        #pragma unroll
        for (int j = 0; j < 8; ++j) {
            const int co = co_base + rowoff + j;
            O[co * (HH * WW) + w] = acc[nt][j];
        }
    }
}

extern "C" void kernel_launch(void* const* d_in, const int* in_sizes, int n_in,
                              void* d_out, int out_size, void* d_ws, size_t ws_size,
                              hipStream_t stream) {
    const float* feat = (const float*)d_in[0];  // (16,128,64,64) f32
    const float* dynk = (const float*)d_in[1];  // (16,128,128,3,3) f32
    float* out = (float*)d_out;                 // (16,128,64,64) f32

    dim3 grid(HH, 16, 1);   // (h row, batch)
    dim3 block(256, 1, 1);  // 8 waves
    dynconv_wmma_bf16_kernel<<<grid, block, 0, stream>>>(feat, dynk, out);
}
